// CrossAttentionSVD_36644660970257
// MI455X (gfx1250) — matrix operand order repaired and checked by reading, not compiled, not run on hardware
//
#include <hip/hip_runtime.h>
#include <hip/hip_bf16.h>
#include <math.h>
#include <stdint.h>

// ---------------------------------------------------------------------------
// CrossAttentionSVD fused for MI455X (gfx1250, wave32, WMMA 16x16x32 bf16)
//   B=16 N=1024 C=1280 H=8 D=160 R=64 -> concat head dim 224, NCAT=1792
// Pipeline:
//   prep:  fp32->bf16 X; pack combined Wq/Wk/Wv (1792x1280), Wo (1280x1792)
//   gemm:  TDM(tensor_load_to_lds) double-buffered -> LDS -> WMMA
//          Q = X Wq^T, K = X Wk^T, Vt = (X Wv^T)^T per (b,h)
//   attn:  flash-style online softmax, head dim 224, keys in 64-blocks
//   gemm:  out = O Wo^T + bias (fp32 out)
// ---------------------------------------------------------------------------

typedef __attribute__((ext_vector_type(16))) __bf16 v16bf;
typedef __attribute__((ext_vector_type(8)))  __bf16 v8bf;
typedef __attribute__((ext_vector_type(8)))  float  v8f;
typedef __attribute__((ext_vector_type(4)))  unsigned int v4u;
typedef __attribute__((ext_vector_type(8)))  int v8i;
typedef __attribute__((ext_vector_type(4)))  int v4i;

#define HEADS 8
#define DIMH  160
#define RANK  64
#define DCAT  224            // 160 + 64 per head
#define NCAT  1792           // 8 * 224
#define CDIM  1280
#define BDIM  16
#define NSEQ  1024
#define MTOT  (BDIM * NSEQ)  // 16384
#define ATT_SCALE 0.07905694150420949f  // 160^-0.5 (applies to both score terms)

// ---- WMMA tile loads, per CDNA5 ISA 16-bit layouts (wave32) ----------------
// A (16x32, MxK): lane l16 = row M; lanes 0-15 hold K {0..7,16..23},
//                 lanes 16-31 hold K {8..15,24..31}  -> two 16B loads
static __device__ __forceinline__ v16bf load_a_tile(const __bf16* base, int hf) {
  const v8bf lo = *(const v8bf*)(base + hf * 8);
  const v8bf hi = *(const v8bf*)(base + 16 + hf * 8);
  v16bf r;
#pragma unroll
  for (int i = 0; i < 8; ++i) { r[i] = lo[i]; r[i + 8] = hi[i]; }
  return r;
}
// B (32x16, KxN): lane l16 = col N; lanes 0-15 hold K {0..7,16..23},
//                 lanes 16-31 hold K {8..15,24..31}  -> two 16B loads
static __device__ __forceinline__ v16bf load_b_tile(const __bf16* base, int hf) {
  const v8bf lo = *(const v8bf*)(base + hf * 8);
  const v8bf hi = *(const v8bf*)(base + 16 + hf * 8);
  v16bf r;
#pragma unroll
  for (int i = 0; i < 8; ++i) { r[i] = lo[i]; r[i + 8] = hi[i]; }
  return r;
}

static __device__ __forceinline__ v8f wmma_bf16(v16bf a, v16bf b, v8f c) {
  return __builtin_amdgcn_wmma_f32_16x16x32_bf16(false, a, false, b,
                                                 (short)0, c, false, false);
}

// ---- TDM: DMA a (tile_rows x 32) bf16 tile (row stride = K elems) into LDS -
// Descriptor per CDNA5 ISA; 2D tile, higher-dim groups
// zeroed (tile_dim2=0 -> unused). 6-arg builtin form (clang-23 lane).
static __device__ __forceinline__ void tdm_load_tile(const __bf16* gbase,
                                                     int K, int rows_total,
                                                     int tile_rows,
                                                     unsigned lds_addr) {
  unsigned long long ga = (unsigned long long)(uintptr_t)gbase;
  v4u g0;
  g0[0] = 1u;                                       // count=1, user descriptor
  g0[1] = lds_addr;                                 // LDS byte address
  g0[2] = (unsigned)(ga & 0xFFFFFFFFu);             // global_addr[31:0]
  g0[3] = (unsigned)((ga >> 32) & 0x01FFFFFFu) | (2u << 30);  // addr[56:32]|type=2
  v8i g1;
  g1[0] = (int)(1u << 16);                          // data_size=1 (2B), mask=0
  g1[1] = (int)(((unsigned)K & 0xFFFFu) << 16);     // tensor_dim0[15:0]
  g1[2] = (int)((((unsigned)K >> 16) & 0xFFFFu) |
                (((unsigned)rows_total & 0xFFFFu) << 16)); // dim0 hi | dim1 lo
  g1[3] = (int)((((unsigned)rows_total >> 16) & 0xFFFFu) |
                (32u << 16));                       // dim1 hi | tile_dim0=32
  g1[4] = tile_rows;                                // tile_dim1 | tile_dim2=0
  g1[5] = K;                                        // tensor_dim0_stride[31:0]
  g1[6] = 0;                                        // stride hi | dim1_stride lo
  g1[7] = 0;
  v4i gz4 = {0, 0, 0, 0};
  v8i gz8 = {0, 0, 0, 0, 0, 0, 0, 0};
  __builtin_amdgcn_tensor_load_to_lds(g0, g1, gz4, gz4, gz8, 0);
}

// ---- prep kernels ----------------------------------------------------------
__global__ void convert_x_kernel(const float* __restrict__ X,
                                 __bf16* __restrict__ Xb, int n) {
  int i = blockIdx.x * blockDim.x + threadIdx.x;
  if (i < n) Xb[i] = (__bf16)X[i];
}

// out (NCAT x CDIM): row h*224+j : j<160 -> Wfull[h*160+j], else Wlr[h*64+j-160]
__global__ void pack_qkv_kernel(const float* __restrict__ Wfull,
                                const float* __restrict__ Wlr,
                                __bf16* __restrict__ out) {
  int i = blockIdx.x * blockDim.x + threadIdx.x;
  if (i >= NCAT * CDIM) return;
  int rr = i / CDIM, c = i % CDIM;
  int h = rr / DCAT, j = rr % DCAT;
  float v = (j < DIMH)
                ? Wfull[(size_t)(h * DIMH + j) * CDIM + c]
                : Wlr[(size_t)(h * RANK + (j - DIMH)) * CDIM + c];  // ALPHA=1
  out[i] = (__bf16)v;
}

// out (CDIM x NCAT): col h*224+j : j<160 -> Wout_full[:,h*160+j] else Wout_lr
__global__ void pack_wo_kernel(const float* __restrict__ Wfull,
                               const float* __restrict__ Wlr,
                               __bf16* __restrict__ out) {
  int i = blockIdx.x * blockDim.x + threadIdx.x;
  if (i >= CDIM * NCAT) return;
  int c = i / NCAT, k = i % NCAT;
  int h = k / DCAT, j = k % DCAT;
  float v = (j < DIMH)
                ? Wfull[(size_t)c * CDIM + h * DIMH + j]
                : Wlr[(size_t)c * (HEADS * RANK) + h * RANK + (j - DIMH)];
  out[i] = (__bf16)v;
}

// ---- GEMM: Y(M,N) = A(M,K) @ W(N,K)^T, bf16 in / fp32 acc ------------------
// 256 threads = 8 waves; block tile 128M x 64N; wave computes 16 rows x 64 cols.
// TDM double-buffers A(128x32) and B(64x32) K-slices into LDS; wave 0 issues
// the DMAs for step i+1, then s_wait_tensorcnt(2) retires step i's pair.
// mode 0: bf16 natural   mode 1: bf16 transposed per-batch (for Vt)
// mode 2: fp32 + bias (final projection)
__global__ __launch_bounds__(256) void gemm_bf16_wmma(
    const __bf16* __restrict__ A, const __bf16* __restrict__ W,
    __bf16* __restrict__ outB, float* __restrict__ outF,
    const float* __restrict__ bias, int M, int N, int K, int mode) {
  __shared__ __attribute__((aligned(16))) __bf16 Al[2][128][32];
  __shared__ __attribute__((aligned(16))) __bf16 Bl[2][64][32];
  const int lane = threadIdx.x & 31;
  const int wave = threadIdx.x >> 5;
  const int l16 = lane & 15;
  const int hf = lane >> 4;
  const int rowBlk = blockIdx.y * 128;
  const int colBase = blockIdx.x * 64;
  const int ksteps = K / 32;

  if (wave == 0) {
    tdm_load_tile(A + (size_t)rowBlk * K, K, M, 128,
                  (unsigned)(uintptr_t)&Al[0][0][0]);
    tdm_load_tile(W + (size_t)colBase * K, K, N, 64,
                  (unsigned)(uintptr_t)&Bl[0][0][0]);
  }

  v8f acc[4] = {};
  for (int i = 0; i < ksteps; ++i) {
    if (wave == 0) {
      if (i + 1 < ksteps) {
        tdm_load_tile(A + (size_t)rowBlk * K + (i + 1) * 32, K, M, 128,
                      (unsigned)(uintptr_t)&Al[(i + 1) & 1][0][0]);
        tdm_load_tile(W + (size_t)colBase * K + (i + 1) * 32, K, N, 64,
                      (unsigned)(uintptr_t)&Bl[(i + 1) & 1][0][0]);
        __builtin_amdgcn_s_wait_tensorcnt(2);  // step i's pair has landed
      } else {
        __builtin_amdgcn_s_wait_tensorcnt(0);
      }
    }
    __syncthreads();  // LDS tiles for step i visible to all waves

    v16bf a = load_a_tile(&Al[i & 1][wave * 16 + l16][0], hf);
#pragma unroll
    for (int t = 0; t < 4; ++t) {
      v16bf b = load_b_tile(&Bl[i & 1][t * 16 + l16][0], hf);
      acc[t] = wmma_bf16(a, b, acc[t]);
    }
    __syncthreads();  // done reading before buffer is overwritten next step
  }

#pragma unroll
  for (int t = 0; t < 4; ++t) {
#pragma unroll
    for (int r = 0; r < 8; ++r) {
      int mrow = rowBlk + wave * 16 + r + hf * 8;  // C/D: M split by lane half
      int ncol = colBase + t * 16 + l16;           // N striped across lanes
      float v = acc[t][r];
      if (mode == 0) {
        outB[(size_t)mrow * N + ncol] = (__bf16)v;
      } else if (mode == 1) {
        int bb = mrow >> 10, tok = mrow & 1023;  // Vt[(b*NCAT+n)*NSEQ+tok]
        outB[((size_t)bb * NCAT + ncol) * NSEQ + tok] = (__bf16)v;
      } else {
        outF[(size_t)mrow * N + ncol] = v + bias[ncol];
      }
    }
  }
}

// ---- flash attention: head dim 224, online softmax over 64-key blocks ------
// 128 threads = 4 waves; wave owns 16 query rows. grid = (NSEQ/64, B*H).
__global__ __launch_bounds__(128) void attn_wmma(
    const __bf16* __restrict__ Q, const __bf16* __restrict__ Kmat,
    const __bf16* __restrict__ Vt, __bf16* __restrict__ O) {
  __shared__ __attribute__((aligned(16))) __bf16 Pl[4][16][64];  // P per wave
  const int lane = threadIdx.x & 31;
  const int wave = threadIdx.x >> 5;
  const int l16 = lane & 15;
  const int hf = lane >> 4;
  const int bh = blockIdx.y;  // b*8 + h
  const int b = bh >> 3;
  const int h = bh & 7;
  const int q0 = blockIdx.x * 64 + wave * 16;

  // preload this wave's 16 query rows as 7 A-tiles (K = 224 = 7*32)
  const __bf16* qrow = Q + ((size_t)(b * NSEQ + q0 + l16)) * NCAT + h * DCAT;
  v16bf aq[7];
#pragma unroll
  for (int ks = 0; ks < 7; ++ks) aq[ks] = load_a_tile(qrow + ks * 32, hf);

  v8f o[14] = {};
  float mr[8], lr[8];
#pragma unroll
  for (int r = 0; r < 8; ++r) { mr[r] = -1e30f; lr[r] = 0.0f; }

  for (int kb = 0; kb < NSEQ; kb += 64) {
    // prefetch next key block's rows (global_prefetch_b8)
    if (kb + 64 < NSEQ) {
      __builtin_prefetch(
          Kmat + ((size_t)(b * NSEQ + kb + 64 + l16)) * NCAT + h * DCAT, 0, 1);
    }
    // S = Q @ K^T  (16q x 64k per wave, 4 tiles x 7 k-steps)
    v8f s[4] = {};
#pragma unroll
    for (int t = 0; t < 4; ++t) {
      const __bf16* krow =
          Kmat + ((size_t)(b * NSEQ + kb + t * 16 + l16)) * NCAT + h * DCAT;
#pragma unroll
      for (int ks = 0; ks < 7; ++ks) {
        v16bf bK = load_b_tile(krow + ks * 32, hf);
        s[t] = wmma_bf16(aq[ks], bK, s[t]);
      }
    }
#pragma unroll
    for (int t = 0; t < 4; ++t)
#pragma unroll
      for (int r = 0; r < 8; ++r) s[t][r] *= ATT_SCALE;

    // online softmax: row = r (lanes 0-15) / r+8 (lanes 16-31); N in lanes.
    // xor masks 1/2/4/8 reduce within each 16-lane half -> per-row reductions.
#pragma unroll
    for (int r = 0; r < 8; ++r) {
      float mx = fmaxf(fmaxf(s[0][r], s[1][r]), fmaxf(s[2][r], s[3][r]));
      mx = fmaxf(mx, __shfl_xor(mx, 1));
      mx = fmaxf(mx, __shfl_xor(mx, 2));
      mx = fmaxf(mx, __shfl_xor(mx, 4));
      mx = fmaxf(mx, __shfl_xor(mx, 8));
      float nm = fmaxf(mr[r], mx);
      float sc = __expf(mr[r] - nm);
      mr[r] = nm;
      float rs = 0.0f;
#pragma unroll
      for (int t = 0; t < 4; ++t) {
        float p = __expf(s[t][r] - nm);
        s[t][r] = p;
        rs += p;
      }
      rs += __shfl_xor(rs, 1);
      rs += __shfl_xor(rs, 2);
      rs += __shfl_xor(rs, 4);
      rs += __shfl_xor(rs, 8);
      lr[r] = lr[r] * sc + rs;
#pragma unroll
      for (int t = 0; t < 14; ++t) o[t][r] *= sc;
    }

    // P (C/D layout) -> LDS bf16 so it can be re-read in A layout
#pragma unroll
    for (int t = 0; t < 4; ++t)
#pragma unroll
      for (int r = 0; r < 8; ++r)
        Pl[wave][r + hf * 8][t * 16 + l16] = (__bf16)s[t][r];
    __syncthreads();

    // O += P @ V   (Vt rows are contiguous over keys -> contiguous B tiles)
#pragma unroll
    for (int t = 0; t < 14; ++t) {
      const __bf16* vrow =
          Vt + ((size_t)(bh * DCAT + t * 16 + l16)) * NSEQ + kb;
#pragma unroll
      for (int k2 = 0; k2 < 2; ++k2) {
        v16bf aP = load_a_tile(&Pl[wave][l16][k2 * 32], hf);
        v16bf bV = load_b_tile(vrow + k2 * 32, hf);
        o[t] = wmma_bf16(aP, bV, o[t]);
      }
    }
    __syncthreads();
  }

  // normalize by softmax denominator and store bf16 O (M x NCAT)
#pragma unroll
  for (int r = 0; r < 8; ++r) {
    float inv = 1.0f / lr[r];
    int row = q0 + r + hf * 8;
    __bf16* orow = O + ((size_t)(b * NSEQ + row)) * NCAT + h * DCAT;
#pragma unroll
    for (int t = 0; t < 14; ++t) orow[t * 16 + l16] = (__bf16)(o[t][r] * inv);
  }
}

// ---------------------------------------------------------------------------
extern "C" void kernel_launch(void* const* d_in, const int* in_sizes, int n_in,
                              void* d_out, int out_size, void* d_ws,
                              size_t ws_size, hipStream_t stream) {
  (void)in_sizes; (void)n_in; (void)out_size; (void)ws_size;
  const float* X         = (const float*)d_in[0];
  const float* Wq_lr     = (const float*)d_in[1];
  const float* Wk_lr     = (const float*)d_in[2];
  const float* Wv_lr     = (const float*)d_in[3];
  const float* Wout_lr   = (const float*)d_in[4];
  const float* Wq_full   = (const float*)d_in[5];
  const float* Wk_full   = (const float*)d_in[6];
  const float* Wv_full   = (const float*)d_in[7];
  const float* Wout_full = (const float*)d_in[8];
  const float* b_out     = (const float*)d_in[9];

  char* ws = (char*)d_ws;
  size_t off = 0;
  auto alloc = [&](size_t bytes) {
    char* p = ws + off;
    off += (bytes + 255) & ~(size_t)255;
    return p;
  };
  __bf16* Xb  = (__bf16*)alloc((size_t)MTOT * CDIM * 2);
  __bf16* Wq  = (__bf16*)alloc((size_t)NCAT * CDIM * 2);
  __bf16* Wk  = (__bf16*)alloc((size_t)NCAT * CDIM * 2);
  __bf16* Wv  = (__bf16*)alloc((size_t)NCAT * CDIM * 2);
  __bf16* Wo  = (__bf16*)alloc((size_t)CDIM * NCAT * 2);
  __bf16* Qb  = (__bf16*)alloc((size_t)MTOT * NCAT * 2);
  __bf16* Kb  = (__bf16*)alloc((size_t)MTOT * NCAT * 2);
  __bf16* Vtb = (__bf16*)alloc((size_t)MTOT * NCAT * 2);
  __bf16* Ob  = (__bf16*)alloc((size_t)MTOT * NCAT * 2);

  const int nX = MTOT * CDIM;
  convert_x_kernel<<<(nX + 255) / 256, 256, 0, stream>>>(X, Xb, nX);
  const int nW = NCAT * CDIM;
  pack_qkv_kernel<<<(nW + 255) / 256, 256, 0, stream>>>(Wq_full, Wq_lr, Wq);
  pack_qkv_kernel<<<(nW + 255) / 256, 256, 0, stream>>>(Wk_full, Wk_lr, Wk);
  pack_qkv_kernel<<<(nW + 255) / 256, 256, 0, stream>>>(Wv_full, Wv_lr, Wv);
  pack_wo_kernel<<<(nW + 255) / 256, 256, 0, stream>>>(Wout_full, Wout_lr, Wo);

  dim3 gQ(NCAT / 64, MTOT / 128);
  gemm_bf16_wmma<<<gQ, 256, 0, stream>>>(Xb, Wq, Qb, nullptr, nullptr,
                                         MTOT, NCAT, CDIM, 0);
  gemm_bf16_wmma<<<gQ, 256, 0, stream>>>(Xb, Wk, Kb, nullptr, nullptr,
                                         MTOT, NCAT, CDIM, 0);
  gemm_bf16_wmma<<<gQ, 256, 0, stream>>>(Xb, Wv, Vtb, nullptr, nullptr,
                                         MTOT, NCAT, CDIM, 1);

  dim3 gA(NSEQ / 64, BDIM * HEADS);
  attn_wmma<<<gA, 128, 0, stream>>>(Qb, Kb, Vtb, Ob);

  dim3 gO(CDIM / 64, MTOT / 128);
  gemm_bf16_wmma<<<gO, 256, 0, stream>>>(Ob, Wo, nullptr, (float*)d_out, b_out,
                                         MTOT, CDIM, NCAT, 2);
}
